// MyGNN_35596688949519
// MI455X (gfx1250) — matrix operand order repaired and checked by reading, not compiled, not run on hardware
//
#include <hip/hip_runtime.h>
#include <hip/hip_bf16.h>
#include <math.h>

// Problem constants (match reference)
#define B_GR   16
#define N_NODE 1024
#define NT     (B_GR * N_NODE)   // 16384 total nodes
#define F_IN   256
#define H_HEAD 4
#define C_CH   64
#define HID    (H_HEAD * C_CH)   // 256
#define L_LAY  3
#define NEG_SLOPE 0.2f

typedef _Float16 v16h __attribute__((ext_vector_type(16)));
typedef _Float16 v8h  __attribute__((ext_vector_type(8)));
typedef float    v8f  __attribute__((ext_vector_type(8)));

// ---------------------------------------------------------------- utilities
__global__ void fill_f32(float* __restrict__ p, float v, int n) {
    int i = blockIdx.x * blockDim.x + threadIdx.x;
    if (i < n) p[i] = v;
}

__global__ void cvt_f32_to_f16(const float* __restrict__ s, _Float16* __restrict__ d, int n) {
    int i = blockIdx.x * blockDim.x + threadIdx.x;
    if (i < n) d[i] = (_Float16)s[i];
}

// W layout: [K=256][N=256] (f, h*C+c). Produce WT[N][K] in f16 so B-fragment
// loads are contiguous along K.
__global__ void cvt_w_T(const float* __restrict__ W, _Float16* __restrict__ WT) {
    int i = blockIdx.x * blockDim.x + threadIdx.x;     // i = n*256 + k
    if (i >= F_IN * HID) return;
    int n = i >> 8, k = i & 255;
    WT[i] = (_Float16)W[k * HID + n];
}

// ---------------------------------------------------------------- WMMA GEMM
// C[M][256] = A[M][256] * W[256][256], A in f16 row-major, W given as WT[N][K].
// One wave32 computes one 16x16 tile; K-loop of 8x v_wmma_f32_16x16x32_f16.
//
// Fragment layouts per CDNA5 ISA 7.12.2 (wave32):
//   A (16x32 f16):  lane half = lane>>4, row m = lane&15,
//                   halves e=0..7  -> K = 8*half + e        (contiguous 16B)
//                   halves e=8..15 -> K = 16 + 8*half + (e-8) (contiguous 16B)
//   B (32x16 f16):  col n = lane&15, halves e=0..15 -> same K order as A
//   C/D (16x16 f32): VGPR r -> row = r + 8*half, col = lane&15
__global__ void __launch_bounds__(256)
gat_gemm_wmma(const _Float16* __restrict__ A, const _Float16* __restrict__ BT,
              float* __restrict__ Cm, int M) {
    const int K = 256, N = 256;
    int wave = (blockIdx.x * blockDim.x + threadIdx.x) >> 5;
    int lane = threadIdx.x & 31;
    const int ntiles = N / 16;
    int tm = wave / ntiles;
    int tn = wave - tm * ntiles;
    if (tm * 16 >= M) return;                 // wave-uniform (EXEC stays all-1s)

    int half = lane >> 4;
    int r    = lane & 15;
    const _Float16* Arow = A  + (size_t)(tm * 16 + r) * K;
    const _Float16* Brow = BT + (size_t)(tn * 16 + r) * K;

    v8f acc = {};
#pragma unroll
    for (int k0 = 0; k0 < K; k0 += 32) {
        v8h alo = *(const v8h*)(Arow + k0 + 8 * half);
        v8h ahi = *(const v8h*)(Arow + k0 + 16 + 8 * half);
        v8h blo = *(const v8h*)(Brow + k0 + 8 * half);
        v8h bhi = *(const v8h*)(Brow + k0 + 16 + 8 * half);
        v16h a, b;
#pragma unroll
        for (int e = 0; e < 8; ++e) {
            a[e] = alo[e]; a[e + 8] = ahi[e];
            b[e] = blo[e]; b[e + 8] = bhi[e];
        }
        acc = __builtin_amdgcn_wmma_f32_16x16x32_f16(
            /*neg_a=*/false, a, /*neg_b=*/false, b,
            /*c_mod=*/(short)0, acc, /*reuse_a=*/false, /*reuse_b=*/false);
    }

    float* Crow = Cm + (size_t)(tm * 16 + 8 * half) * N + tn * 16 + r;
#pragma unroll
    for (int rr = 0; rr < 8; ++rr) Crow[(size_t)rr * N] = acc[rr];
}

// ---------------------------------------------------------------- attention
// s[n,h] = sum_c xp[n,h,c]*a_src[h,c];  d likewise.
__global__ void attn_coef(const float* __restrict__ xp,
                          const float* __restrict__ a_src,
                          const float* __restrict__ a_dst,
                          float* __restrict__ s, float* __restrict__ d) {
    int i = blockIdx.x * blockDim.x + threadIdx.x;   // n*H + h
    if (i >= NT * H_HEAD) return;
    int n = i >> 2, h = i & 3;
    const float* xr = xp + (size_t)n * HID + h * C_CH;
    const float* as = a_src + h * C_CH;
    const float* ad = a_dst + h * C_CH;
    float ss = 0.f, dd = 0.f;
#pragma unroll 4
    for (int c = 0; c < C_CH; ++c) { float v = xr[c]; ss += v * as[c]; dd += v * ad[c]; }
    s[i] = ss; d[i] = dd;
}

// order-preserving float atomic max (buffer pre-filled with -inf)
__device__ __forceinline__ void atomicMaxF(float* addr, float v) {
    if (v >= 0.f) atomicMax((int*)addr, __float_as_int(v));
    else          atomicMin((unsigned int*)addr, __float_as_uint(v));
}

__device__ __forceinline__ void edge_endpoints(int eh, int E,
                                               const int* __restrict__ src,
                                               const int* __restrict__ dst,
                                               int& si, int& di) {
    if (eh < E) { si = src[eh]; di = dst[eh]; }
    else        { si = di = eh - E; }          // implicit self-loops
}

__global__ void edge_max(const int* __restrict__ src, const int* __restrict__ dst,
                         const float* __restrict__ s, const float* __restrict__ d,
                         float* __restrict__ maxb, int E) {
    int t = blockIdx.x * blockDim.x + threadIdx.x;
    if (t >= (E + NT) * H_HEAD) return;
    int eh = t >> 2, h = t & 3, si, di;
    edge_endpoints(eh, E, src, dst, si, di);
    float e = s[si * H_HEAD + h] + d[di * H_HEAD + h];
    e = e > 0.f ? e : NEG_SLOPE * e;
    atomicMaxF(&maxb[di * H_HEAD + h], e);
}

__global__ void edge_expsum(const int* __restrict__ src, const int* __restrict__ dst,
                            const float* __restrict__ s, const float* __restrict__ d,
                            const float* __restrict__ maxb,
                            float* __restrict__ exb, float* __restrict__ sumb, int E) {
    int t = blockIdx.x * blockDim.x + threadIdx.x;
    if (t >= (E + NT) * H_HEAD) return;
    int eh = t >> 2, h = t & 3, si, di;
    edge_endpoints(eh, E, src, dst, si, di);
    float e = s[si * H_HEAD + h] + d[di * H_HEAD + h];
    e = e > 0.f ? e : NEG_SLOPE * e;
    float ex = __expf(e - maxb[di * H_HEAD + h]);
    exb[t] = ex;
    atomicAdd(&sumb[di * H_HEAD + h], ex);
}

// one 256-thread block per edge; thread hc handles channel (h = hc>>6)
__global__ void __launch_bounds__(256)
edge_scatter(const int* __restrict__ src, const int* __restrict__ dst,
             const float* __restrict__ exb, const float* __restrict__ sumb,
             const float* __restrict__ xp, float* __restrict__ outa, int E) {
    int eh = blockIdx.x;
    int hc = threadIdx.x;
    int si, di;
    edge_endpoints(eh, E, src, dst, si, di);
    int h = hc >> 6;
    float alpha = exb[(size_t)eh * H_HEAD + h] / sumb[di * H_HEAD + h];
    atomicAdd(&outa[(size_t)di * HID + hc], alpha * xp[(size_t)si * HID + hc]);
}

__global__ void finalize(const float* __restrict__ outa, const float* __restrict__ bias,
                         float* __restrict__ xnext, float* __restrict__ out, int do_relu) {
    int i = blockIdx.x * blockDim.x + threadIdx.x;
    if (i >= NT * HID) return;
    float v = outa[i] + bias[i & (HID - 1)];
    if (do_relu) { xnext[i] = fmaxf(v, 0.f); }
    else         { out[i]   = v; }
}

// ---------------------------------------------------------------- launcher
extern "C" void kernel_launch(void* const* d_in, const int* in_sizes, int n_in,
                              void* d_out, int out_size, void* d_ws, size_t ws_size,
                              hipStream_t stream) {
    const float* node_feature = (const float*)d_in[0];           // [B,N,F]
    const float* Ws           = (const float*)d_in[1];           // [L,F,H,C]
    const float* att_src      = (const float*)d_in[2];           // [L,H,C]
    const float* att_dst      = (const float*)d_in[3];           // [L,H,C]
    const float* bias         = (const float*)d_in[4];           // [L,HID]
    const int*   ei           = (const int*)d_in[5];             // [2,E]
    const int E  = in_sizes[5] / 2;
    const int ET = E + NT;
    const int* src = ei;
    const int* dst = ei + E;

    // carve workspace (~62 MB total)
    char* w = (char*)d_ws;
    auto carve = [&](size_t bytes) {
        char* p = w; w += (bytes + 255) & ~(size_t)255; return (void*)p;
    };
    float*    x_cur = (float*)   carve((size_t)NT * HID * 4);
    _Float16* xh    = (_Float16*)carve((size_t)NT * HID * 2);
    _Float16* WT    = (_Float16*)carve((size_t)F_IN * HID * 2);
    float*    xp    = (float*)   carve((size_t)NT * HID * 4);
    float*    sbuf  = (float*)   carve((size_t)NT * H_HEAD * 4);
    float*    dbuf  = (float*)   carve((size_t)NT * H_HEAD * 4);
    float*    maxb  = (float*)   carve((size_t)NT * H_HEAD * 4);
    float*    sumb  = (float*)   carve((size_t)NT * H_HEAD * 4);
    float*    exb   = (float*)   carve((size_t)ET * H_HEAD * 4);
    float*    outa  = (float*)   carve((size_t)NT * HID * 4);

    const int TB = 256;
    auto gs = [&](int n) { return (n + TB - 1) / TB; };

    for (int l = 0; l < L_LAY; ++l) {
        const float* xin = (l == 0) ? node_feature : x_cur;

        cvt_f32_to_f16<<<gs(NT * HID), TB, 0, stream>>>(xin, xh, NT * HID);
        cvt_w_T<<<gs(F_IN * HID), TB, 0, stream>>>(Ws + (size_t)l * F_IN * HID, WT);

        // (NT/16)*(HID/16) = 16384 waves -> 2048 blocks of 8 waves
        gat_gemm_wmma<<<(NT / 16) * (HID / 16) * 32 / TB, TB, 0, stream>>>(xh, WT, xp, NT);

        attn_coef<<<gs(NT * H_HEAD), TB, 0, stream>>>(
            xp, att_src + (size_t)l * HID, att_dst + (size_t)l * HID, sbuf, dbuf);

        fill_f32<<<gs(NT * H_HEAD), TB, 0, stream>>>(maxb, -INFINITY, NT * H_HEAD);
        fill_f32<<<gs(NT * H_HEAD), TB, 0, stream>>>(sumb, 0.f, NT * H_HEAD);
        fill_f32<<<gs(NT * HID),    TB, 0, stream>>>(outa, 0.f, NT * HID);

        edge_max<<<gs(ET * H_HEAD), TB, 0, stream>>>(src, dst, sbuf, dbuf, maxb, E);
        edge_expsum<<<gs(ET * H_HEAD), TB, 0, stream>>>(src, dst, sbuf, dbuf, maxb, exb, sumb, E);
        edge_scatter<<<ET, TB, 0, stream>>>(src, dst, exb, sumb, xp, outa, E);

        finalize<<<gs(NT * HID), TB, 0, stream>>>(
            outa, bias + (size_t)l * HID, x_cur, (float*)d_out, l < L_LAY - 1);
    }
}
